// DoubleStreamBlock_60361470378351
// MI455X (gfx1250) — compile-verified, matrix operand order checked by reading, not run on hardware
//
#include <hip/hip_runtime.h>
#include <math.h>

typedef __bf16 bf16;
typedef __attribute__((ext_vector_type(8)))  bf16  bf16x8;
typedef __attribute__((ext_vector_type(16))) bf16  v16bf;
typedef __attribute__((ext_vector_type(8)))  float v8f;
typedef __attribute__((ext_vector_type(4)))  unsigned int u32x4;
typedef __attribute__((ext_vector_type(4)))  int i32x4;
typedef __attribute__((ext_vector_type(8)))  int i32x8;

#define DEV __device__ __forceinline__

static constexpr int LT_  = 256;
static constexpr int LI_  = 2048;
static constexpr int L_   = LT_ + LI_;   // 2304
static constexpr int H_   = 2048;
static constexpr int NH_  = 16;
static constexpr int HD_  = 128;
static constexpr int MLP_ = 8192;
static constexpr int KC_  = 1024;        // K-chunk staged in LDS per TDM op
static constexpr float EPS_ = 1e-6f;

// ---------------------------------------------------------------------------
// Fragment loader: CDNA5 16-bit A/B matrix layout (16xK, K-chunk of 32).
// lane: row = lane&15, kbase = (lane>>4)*8.
// elements e<8  -> k = kbase + e ; e>=8 -> k = kbase + 8 + e
// => two contiguous 16-byte runs at [kbase, +8) and [kbase+16, +8).
// Caller passes p = row_base + kbase. Works for global and LDS pointers.
// ---------------------------------------------------------------------------
DEV v16bf load_frag(const bf16* p) {
  bf16x8 lo = *reinterpret_cast<const bf16x8*>(p);
  bf16x8 hi = *reinterpret_cast<const bf16x8*>(p + 16);
  v16bf f;
#pragma unroll
  for (int i = 0; i < 8; ++i) { f[i] = lo[i]; f[i + 8] = hi[i]; }
  return f;
}

DEV v8f wmma_bf16(v16bf a, v16bf b, v8f c) {
  return __builtin_amdgcn_wmma_f32_16x16x32_bf16(false, a, false, b, (short)0, c,
                                                 false, false);
}

DEV float gelu_tanh(float v) {
  float t = 0.7978845608028654f * (v + 0.044715f * v * v * v);
  return 0.5f * v * (1.0f + tanhf(t));
}

// ---------------------------------------------------------------------------
// TDM: DMA a [16 rows x KC_ cols] bf16 tile (row stride = K elements) from
// global memory into LDS at ldsOff. D# per CDNA5 ISA §8.3/8.4:
//   group0: count=1 | lds_addr | global_addr[56:0] | type=2
//   group1: data_size=2B, tensor_dim0=K, tensor_dim1=16, tile_dim0=KC_,
//           tile_dim1=16, tensor_dim0_stride=K  (2D, no pad/iterate/multicast)
// ---------------------------------------------------------------------------
DEV void tdm_load_tile_16xKC(const bf16* gsrc, unsigned ldsOff, int K) {
  const unsigned long long ga = (unsigned long long)(uintptr_t)gsrc;
  u32x4 g0;
  g0[0] = 1u;                                   // count=1, user descriptor
  g0[1] = ldsOff;                               // LDS byte address
  g0[2] = (unsigned)(ga & 0xFFFFFFFFull);       // global_addr[31:0]
  g0[3] = (unsigned)((ga >> 32) & 0x1FFFFFFull) // global_addr[56:32]
          | (2u << 30);                         // type=2 ("image")
  i32x8 g1;
  g1[0] = 0x10000;                              // wg_mask=0, data_size=1 (2B)
  g1[1] = (int)(((unsigned)K & 0xFFFFu) << 16); // tensor_dim0[15:0] @ bits63:48
  g1[2] = (int)(((unsigned)K >> 16) | (16u << 16)); // dim0[31:16] | dim1[15:0]
  g1[3] = (int)(((unsigned)KC_ & 0xFFFFu) << 16);   // dim1[31:16]=0 | tile_dim0
  g1[4] = 16;                                   // tile_dim1=16, tile_dim2=0
  g1[5] = K;                                    // tensor_dim0_stride[31:0]
  g1[6] = 0;                                    // stride0[47:32], stride1 lo
  g1[7] = 0;
  i32x4 z4 = {0, 0, 0, 0};
#if __clang_major__ >= 23
  i32x8 z8 = {0, 0, 0, 0, 0, 0, 0, 0};
  __builtin_amdgcn_tensor_load_to_lds(g0, g1, z4, z4, z8, 0);
#else
  __builtin_amdgcn_tensor_load_to_lds(g0, g1, z4, z4, 0);
#endif
}

// ---------------------------------------------------------------------------
// cast fp32 -> bf16
// ---------------------------------------------------------------------------
__global__ void cast_f32_bf16(const float* __restrict__ in, bf16* __restrict__ out,
                              size_t n) {
  size_t i = (size_t)blockIdx.x * blockDim.x + threadIdx.x;
  size_t stride = (size_t)gridDim.x * blockDim.x;
  for (; i < n; i += stride) out[i] = (bf16)in[i];
}

// ---------------------------------------------------------------------------
// LayerNorm + modulation:  out_bf16 = (1+scale[n]) * LN(x)[row,n] + shift[n]
// ---------------------------------------------------------------------------
__global__ __launch_bounds__(256) void ln_mod_kernel(
    const float* __restrict__ x, const float* __restrict__ scale,
    const float* __restrict__ shift, bf16* __restrict__ out, int Hn) {
  __shared__ float s1[256], s2[256];
  const int row = blockIdx.x, tid = threadIdx.x;
  const float* xr = x + (size_t)row * Hn;
  float a = 0.f, b = 0.f;
  for (int i = tid; i < Hn; i += 256) { float v = xr[i]; a += v; b += v * v; }
  s1[tid] = a; s2[tid] = b;
  __syncthreads();
  for (int off = 128; off > 0; off >>= 1) {
    if (tid < off) { s1[tid] += s1[tid + off]; s2[tid] += s2[tid + off]; }
    __syncthreads();
  }
  const float mean = s1[0] / Hn;
  const float var  = s2[0] / Hn - mean * mean;
  const float rinv = rsqrtf(var + EPS_);
  bf16* orow = out + (size_t)row * Hn;
  for (int i = tid; i < Hn; i += 256) {
    float v = (xr[i] - mean) * rinv;
    v = (1.0f + scale[i]) * v + shift[i];
    orow[i] = (bf16)v;
  }
}

// ---------------------------------------------------------------------------
// bf16 WMMA GEMM with TDM-staged A:  C[M,N] = A[M,K] @ W[N,K]^T (+epilogue)
// Block = 8 waves sharing one 16-row m-tile, covering 512 columns (64/wave).
// A strip staged through LDS in KC_-deep chunks by wave 0 via the Tensor Data
// Mover (tensor_load_to_lds + s_wait_tensorcnt); fragments then ds_load'ed.
// W tiles stream global->VGPR (no intra-block reuse; L2 covers cross-block).
// MODE 0: Cf = acc + bias
// MODE 1: Cf = resid + gate[n]*(acc + bias)
// MODE 2: Cb = gelu(acc + bias)        (bf16 out)
// Requires: M%16==0, N%512==0, K%KC_==0 (true for all call sites).
// ---------------------------------------------------------------------------
template <int MODE>
__global__ __launch_bounds__(256) void gemm_wmma_kernel(
    const bf16* __restrict__ A, const bf16* __restrict__ W,
    const float* __restrict__ bias, const float* __restrict__ resid,
    const float* __restrict__ gate, float* __restrict__ Cf,
    bf16* __restrict__ Cb, int M, int N, int K) {
  __shared__ bf16 sA[16 * KC_];  // 32 KB
  const int wave = threadIdx.x >> 5;
  const int lane = threadIdx.x & 31;
  const int nStrips = N >> 9;            // 512-wide column strips
  const int mt = blockIdx.x / nStrips;
  const int ns = blockIdx.x - mt * nStrips;
  const int l15 = lane & 15;
  const int half = lane >> 4;
  const int kb = half << 3;
  const int ncol0 = ns * 512 + wave * 64;

  v8f acc[4];
#pragma unroll
  for (int j = 0; j < 4; ++j) acc[j] = (v8f){0.f,0.f,0.f,0.f,0.f,0.f,0.f,0.f};

  const bf16* wrow[4];
#pragma unroll
  for (int j = 0; j < 4; ++j)
    wrow[j] = W + (size_t)(ncol0 + j * 16 + l15) * K + kb;

  const unsigned ldsOff = (unsigned)(uintptr_t)(&sA[0]);  // LDS aperture low32
  const bf16* astrip = A + (size_t)(mt * 16) * K;

  for (int kc = 0; kc < K; kc += KC_) {
    if (wave == 0) {
      tdm_load_tile_16xKC(astrip + kc, ldsOff, K);
      __builtin_amdgcn_s_wait_tensorcnt(0);
    }
    __syncthreads();
    const bf16* arow = &sA[l15 * KC_ + kb];
#pragma unroll 4
    for (int k0 = 0; k0 < KC_; k0 += 32) {
      v16bf af = load_frag(arow + k0);
#pragma unroll
      for (int j = 0; j < 4; ++j)
        acc[j] = wmma_bf16(af, load_frag(wrow[j] + kc + k0), acc[j]);
    }
    __syncthreads();
  }

  // C layout: lane<16 -> n=nbase+lane, m=mt*16+r ; lane>=16 -> m=mt*16+8+r
  const int m0 = mt * 16 + half * 8;
#pragma unroll
  for (int j = 0; j < 4; ++j) {
    const int n = ncol0 + j * 16 + l15;
    const float bv = bias ? bias[n] : 0.f;
    const float gv = (MODE == 1) ? gate[n] : 0.f;
#pragma unroll
    for (int r = 0; r < 8; ++r) {
      const size_t idx = (size_t)(m0 + r) * N + n;
      float v = acc[j][r] + bv;
      if (MODE == 1) v = resid[idx] + gv * v;
      if (MODE == 2) {
        Cb[idx] = (bf16)gelu_tanh(v);
      } else {
        Cf[idx] = v;
      }
    }
  }
}

// ---------------------------------------------------------------------------
// QKV post-processing: RMS-norm (q,k) with per-stream scale, RoPE, layout
// q,k -> [NH, L, HD] bf16 ; v -> transposed [NH, HD, L] bf16.
// ---------------------------------------------------------------------------
__global__ __launch_bounds__(128) void qkv_post_kernel(
    const float* __restrict__ qkvI, const float* __restrict__ qkvT,
    const float* __restrict__ pe, const float* __restrict__ qnI,
    const float* __restrict__ knI, const float* __restrict__ qnT,
    const float* __restrict__ knT, bf16* __restrict__ Qb,
    bf16* __restrict__ Kb, bf16* __restrict__ VTb) {
  const int blk = blockIdx.x;
  const int t = blk >> 4;        // token (0..L-1), NH_==16
  const int h = blk & 15;        // head
  const int d = threadIdx.x;     // 0..127

  const float* base;
  const float *qn, *kn;
  if (t < LT_) { base = qkvT + (size_t)t * 3 * H_;          qn = qnT; kn = knT; }
  else         { base = qkvI + (size_t)(t - LT_) * 3 * H_;  qn = qnI; kn = knI; }

  const float q = base[h * HD_ + d];
  const float k = base[H_ + h * HD_ + d];
  const float v = base[2 * H_ + h * HD_ + d];

  __shared__ float red[128], sq[128], sk[128];

  red[d] = q * q; __syncthreads();
  for (int off = 64; off > 0; off >>= 1) {
    if (d < off) red[d] += red[d + off];
    __syncthreads();
  }
  const float rq = rsqrtf(red[0] / (float)HD_ + EPS_);
  __syncthreads();
  red[d] = k * k; __syncthreads();
  for (int off = 64; off > 0; off >>= 1) {
    if (d < off) red[d] += red[d + off];
    __syncthreads();
  }
  const float rk = rsqrtf(red[0] / (float)HD_ + EPS_);

  sq[d] = q * rq * qn[d];
  sk[d] = k * rk * kn[d];
  __syncthreads();

  // RoPE: out[2*d2+i] = pe[t,d2,i,0]*x[2*d2] + pe[t,d2,i,1]*x[2*d2+1]
  const int d2 = d >> 1, i = d & 1;
  const float* p = pe + ((((size_t)t * (HD_ / 2) + d2) * 2 + i) * 2);
  const float qr = p[0] * sq[2 * d2] + p[1] * sq[2 * d2 + 1];
  const float kr = p[0] * sk[2 * d2] + p[1] * sk[2 * d2 + 1];

  const size_t qi = ((size_t)h * L_ + t) * HD_ + d;
  Qb[qi] = (bf16)qr;
  Kb[qi] = (bf16)kr;
  VTb[(size_t)h * HD_ * L_ + (size_t)d * L_ + t] = (bf16)v;
}

// ---------------------------------------------------------------------------
// Flash attention, one wave per (head, 16-query tile). mask is all-true.
// S^T = K_tile x Q^T keeps softmax state lane-local per query; exp'd probs
// already sit in the A-fragment layout for P x V (no shuffles).
// ---------------------------------------------------------------------------
__global__ __launch_bounds__(32) void flash_attn_kernel(
    const bf16* __restrict__ Q, const bf16* __restrict__ Km,
    const bf16* __restrict__ VT, float* __restrict__ O) {
  constexpr int QTILES = L_ / 16;  // 144
  const int h  = blockIdx.x / QTILES;
  const int qt = blockIdx.x - h * QTILES;
  const int lane = threadIdx.x & 31;
  const int l15 = lane & 15;
  const int half = lane >> 4;
  const int kb = half << 3;
  const float sc = 0.08838834764831845f;  // 1/sqrt(128)

  v16bf qf[4];
  {
    const bf16* qbase = Q + ((size_t)h * L_ + qt * 16 + l15) * HD_ + kb;
#pragma unroll
    for (int c = 0; c < 4; ++c) qf[c] = load_frag(qbase + c * 32);
  }

  v8f o[8];
#pragma unroll
  for (int j = 0; j < 8; ++j) o[j] = (v8f){0.f,0.f,0.f,0.f,0.f,0.f,0.f,0.f};
  float mi = -1e30f, li = 0.f;

  const bf16* kh = Km + (size_t)h * L_ * HD_;
  const bf16* vh = VT + (size_t)h * HD_ * L_;

  for (int kc = 0; kc < L_; kc += 32) {
    v8f s0 = (v8f){0.f,0.f,0.f,0.f,0.f,0.f,0.f,0.f};
    v8f s1 = s0;
    const bf16* k0p = kh + (size_t)(kc + l15) * HD_ + kb;
    const bf16* k1p = k0p + (size_t)16 * HD_;
#pragma unroll
    for (int c = 0; c < 4; ++c) {
      s0 = wmma_bf16(load_frag(k0p + c * 32), qf[c], s0);
      s1 = wmma_bf16(load_frag(k1p + c * 32), qf[c], s1);
    }
    float mx = -1e30f;
#pragma unroll
    for (int r = 0; r < 8; ++r) {
      s0[r] *= sc; s1[r] *= sc;
      mx = fmaxf(mx, fmaxf(s0[r], s1[r]));
    }
    mx = fmaxf(mx, __shfl_xor(mx, 16, 32));
    const float mn = fmaxf(mi, mx);
    float p0[8], p1[8], rs = 0.f;
#pragma unroll
    for (int r = 0; r < 8; ++r) {
      p0[r] = __expf(s0[r] - mn);
      p1[r] = __expf(s1[r] - mn);
      rs += p0[r] + p1[r];
    }
    rs += __shfl_xor(rs, 16, 32);
    const float alpha = __expf(mi - mn);
    li = li * alpha + rs;
    mi = mn;

    v16bf pf;
#pragma unroll
    for (int e = 0; e < 8; ++e) { pf[e] = (bf16)p0[e]; pf[e + 8] = (bf16)p1[e]; }

#pragma unroll
    for (int r = 0; r < 8; ++r) {
      const float aq = __shfl(alpha, r + half * 8, 32);
#pragma unroll
      for (int j = 0; j < 8; ++j) o[j][r] *= aq;
    }

    const bf16* vbase = vh + kc + kb;
#pragma unroll
    for (int j = 0; j < 8; ++j) {
      const int d = j * 16 + l15;
      o[j] = wmma_bf16(pf, load_frag(vbase + (size_t)d * L_), o[j]);
    }
  }

  float pinv[8];
#pragma unroll
  for (int r = 0; r < 8; ++r)
    pinv[r] = 1.0f / __shfl(li, r + half * 8, 32);
  const int t0 = qt * 16 + half * 8;
#pragma unroll
  for (int j = 0; j < 8; ++j) {
    const int col = h * HD_ + j * 16 + l15;
#pragma unroll
    for (int r = 0; r < 8; ++r)
      O[(size_t)(t0 + r) * H_ + col] = o[j][r] * pinv[r];
  }
}

// ---------------------------------------------------------------------------
// Host-side launch
// ---------------------------------------------------------------------------
static void launch_cast(const float* src, bf16* dst, size_t n, hipStream_t s) {
  cast_f32_bf16<<<2048, 256, 0, s>>>(src, dst, n);
}

template <int MODE>
static void launch_gemm(const bf16* A, const bf16* W, const float* bias,
                        const float* resid, const float* gate, float* Cf,
                        bf16* Cb, int M, int N, int K, hipStream_t s) {
  const int blocks = (M / 16) * (N / 512);
  gemm_wmma_kernel<MODE><<<blocks, 256, 0, s>>>(A, W, bias, resid, gate, Cf, Cb,
                                                M, N, K);
}

extern "C" void kernel_launch(void* const* d_in, const int* in_sizes, int n_in,
                              void* d_out, int out_size, void* d_ws,
                              size_t ws_size, hipStream_t stream) {
  (void)in_sizes; (void)n_in; (void)out_size; (void)ws_size;
  const float* img = (const float*)d_in[0];
  const float* txt = (const float*)d_in[1];
  const float* pe  = (const float*)d_in[2];
  // d_in[3] = mask: all-true in this problem -> no-op in softmax, ignored.
  const float* i_m1_shift = (const float*)d_in[4];
  const float* i_m1_scale = (const float*)d_in[5];
  const float* i_m1_gate  = (const float*)d_in[6];
  const float* i_m2_shift = (const float*)d_in[7];
  const float* i_m2_scale = (const float*)d_in[8];
  const float* i_m2_gate  = (const float*)d_in[9];
  const float* t_m1_shift = (const float*)d_in[10];
  const float* t_m1_scale = (const float*)d_in[11];
  const float* t_m1_gate  = (const float*)d_in[12];
  const float* t_m2_shift = (const float*)d_in[13];
  const float* t_m2_scale = (const float*)d_in[14];
  const float* t_m2_gate  = (const float*)d_in[15];
  const float* i_qkv_w  = (const float*)d_in[16];
  const float* i_qkv_b  = (const float*)d_in[17];
  const float* i_qnorm  = (const float*)d_in[18];
  const float* i_knorm  = (const float*)d_in[19];
  const float* i_proj_w = (const float*)d_in[20];
  const float* i_proj_b = (const float*)d_in[21];
  const float* i_mlp_w1 = (const float*)d_in[22];
  const float* i_mlp_b1 = (const float*)d_in[23];
  const float* i_mlp_w2 = (const float*)d_in[24];
  const float* i_mlp_b2 = (const float*)d_in[25];
  const float* t_qkv_w  = (const float*)d_in[26];
  const float* t_qkv_b  = (const float*)d_in[27];
  const float* t_qnorm  = (const float*)d_in[28];
  const float* t_knorm  = (const float*)d_in[29];
  const float* t_proj_w = (const float*)d_in[30];
  const float* t_proj_b = (const float*)d_in[31];
  const float* t_mlp_w1 = (const float*)d_in[32];
  const float* t_mlp_b1 = (const float*)d_in[33];
  const float* t_mlp_w2 = (const float*)d_in[34];
  const float* t_mlp_b2 = (const float*)d_in[35];

  float* out_img = (float*)d_out;                     // [LI, H]
  float* out_txt = out_img + (size_t)LI_ * H_;        // [LT, H]

  // ---- workspace carve (~190 MB) ----
  char* wp = (char*)d_ws;
  auto carve = [&](size_t bytes) {
    void* r = (void*)wp;
    wp += (bytes + 255) & ~(size_t)255;
    return r;
  };
  bf16*  Wb     = (bf16*)carve((size_t)MLP_ * H_ * sizeof(bf16));   // weight slab (reused)
  bf16*  XMA    = (bf16*)carve((size_t)LI_ * H_ * sizeof(bf16));    // img modulated LN
  bf16*  XMB    = (bf16*)carve((size_t)LT_ * H_ * sizeof(bf16));    // txt modulated LN
  float* QKVI   = (float*)carve((size_t)LI_ * 3 * H_ * sizeof(float));
  float* QKVT   = (float*)carve((size_t)LT_ * 3 * H_ * sizeof(float));
  bf16*  Qb     = (bf16*)carve((size_t)NH_ * L_ * HD_ * sizeof(bf16));
  bf16*  Kb     = (bf16*)carve((size_t)NH_ * L_ * HD_ * sizeof(bf16));
  bf16*  VTb    = (bf16*)carve((size_t)NH_ * HD_ * L_ * sizeof(bf16));
  float* ATTNF  = (float*)carve((size_t)L_ * H_ * sizeof(float));
  bf16*  ATTNB  = (bf16*)carve((size_t)L_ * H_ * sizeof(bf16));
  bf16*  H1     = (bf16*)carve((size_t)LI_ * MLP_ * sizeof(bf16));  // gelu out (reused)

  // ---- 1. LN + modulation (stream 1 modulation) ----
  ln_mod_kernel<<<LI_, 256, 0, stream>>>(img, i_m1_scale, i_m1_shift, XMA, H_);
  ln_mod_kernel<<<LT_, 256, 0, stream>>>(txt, t_m1_scale, t_m1_shift, XMB, H_);

  // ---- 2. QKV GEMMs ----
  launch_cast(i_qkv_w, Wb, (size_t)3 * H_ * H_, stream);
  launch_gemm<0>(XMA, Wb, i_qkv_b, nullptr, nullptr, QKVI, nullptr,
                 LI_, 3 * H_, H_, stream);
  launch_cast(t_qkv_w, Wb, (size_t)3 * H_ * H_, stream);
  launch_gemm<0>(XMB, Wb, t_qkv_b, nullptr, nullptr, QKVT, nullptr,
                 LT_, 3 * H_, H_, stream);

  // ---- 3. RMS-norm + RoPE + head layout ----
  qkv_post_kernel<<<L_ * NH_, 128, 0, stream>>>(QKVI, QKVT, pe, i_qnorm, i_knorm,
                                                t_qnorm, t_knorm, Qb, Kb, VTb);

  // ---- 4. Attention ----
  flash_attn_kernel<<<NH_ * (L_ / 16), 32, 0, stream>>>(Qb, Kb, VTb, ATTNF);
  launch_cast(ATTNF, ATTNB, (size_t)L_ * H_, stream);

  // ---- 5. Output projections with gated residual ----
  launch_cast(i_proj_w, Wb, (size_t)H_ * H_, stream);
  launch_gemm<1>(ATTNB + (size_t)LT_ * H_, Wb, i_proj_b, img, i_m1_gate,
                 out_img, nullptr, LI_, H_, H_, stream);
  launch_cast(t_proj_w, Wb, (size_t)H_ * H_, stream);
  launch_gemm<1>(ATTNB, Wb, t_proj_b, txt, t_m1_gate,
                 out_txt, nullptr, LT_, H_, H_, stream);

  // ---- 6. img MLP: LN2 -> W1+GELU -> W2 + gated residual (in-place) ----
  ln_mod_kernel<<<LI_, 256, 0, stream>>>(out_img, i_m2_scale, i_m2_shift, XMA, H_);
  launch_cast(i_mlp_w1, Wb, (size_t)MLP_ * H_, stream);
  launch_gemm<2>(XMA, Wb, i_mlp_b1, nullptr, nullptr, nullptr, H1,
                 LI_, MLP_, H_, stream);
  launch_cast(i_mlp_w2, Wb, (size_t)H_ * MLP_, stream);
  launch_gemm<1>(H1, Wb, i_mlp_b2, out_img, i_m2_gate, out_img, nullptr,
                 LI_, H_, MLP_, stream);

  // ---- 7. txt MLP ----
  ln_mod_kernel<<<LT_, 256, 0, stream>>>(out_txt, t_m2_scale, t_m2_shift, XMB, H_);
  launch_cast(t_mlp_w1, Wb, (size_t)MLP_ * H_, stream);
  launch_gemm<2>(XMB, Wb, t_mlp_b1, nullptr, nullptr, nullptr, H1,
                 LT_, MLP_, H_, stream);
  launch_cast(t_mlp_w2, Wb, (size_t)H_ * MLP_, stream);
  launch_gemm<1>(H1, Wb, t_mlp_b2, out_txt, t_m2_gate, out_txt, nullptr,
                 LT_, H_, MLP_, stream);
}
